// LGEB_89833535963776
// MI455X (gfx1250) — matrix operand order repaired and checked by reading, not compiled, not run on hardware
//
#include <hip/hip_runtime.h>
#include <hip/hip_bf16.h>

// ---------------------------------------------------------------------------
// LGEB (LorentzNet block) for MI455X / gfx1250.
// Strategy: all GEMMs via v_wmma_f32_16x16x32_f16 (f16 in, f32 accum),
// one wave32 computes one 16-row tile, segment-sums via f32 global atomics.
// ---------------------------------------------------------------------------

typedef __attribute__((ext_vector_type(16))) _Float16 v16h;
typedef __attribute__((ext_vector_type(8)))  float    v8f;

#define CF   96     // C_FEAT == HID
#define KP_E 224    // edge layer-1 K padded: 194 -> 7*32
#define D_H  192    // node layer-1 K: 2*96 = 6*32

union Frag { uint4 u[2]; v16h h; };

// Wave-level LDS producer->consumer sync (per-wave buffers, no block barrier).
__device__ __forceinline__ void lds_sync() {
  __builtin_amdgcn_wave_barrier();
  asm volatile("s_wait_dscnt 0x0" ::: "memory");
  __builtin_amdgcn_wave_barrier();
}

__device__ __forceinline__ float psi_f(float v) {
  return copysignf(log1pf(fabsf(v)), v);
}

// One wave computes OUT(16x96) = relu( A(16,K) @ Bt^T + bias ), OUT -> LDS f16.
// A: LDS f16, row-major, stride astride (halves). Bt: global f16, N-major
// (Bt[n*bstride + k]), so each lane's B fragment (N = lane%16, 8 K at 8*(lane/16)
// and 8 at 16+8*(lane/16)) is two contiguous 16B loads.
template<int KSLABS>
__device__ __forceinline__ void wave_gemm_relu_lds(
    const _Float16* __restrict__ A, int astride,
    const _Float16* __restrict__ Bt, int bstride,
    const float* __restrict__ bias,
    _Float16* __restrict__ out, int lane)
{
  const int row = lane & 15;
  const int hh  = lane >> 4;
#pragma unroll
  for (int nt = 0; nt < 6; ++nt) {
    v8f acc = {};
    const int n = nt * 16 + row;
    const _Float16* bprow = Bt + (size_t)n * bstride;
#pragma unroll
    for (int ks = 0; ks < KSLABS; ++ks) {
      Frag a, b;
      const _Float16* ap = A + row * astride + ks * 32 + hh * 8;
      a.u[0] = *(const uint4*)(ap);
      a.u[1] = *(const uint4*)(ap + 16);
      const _Float16* bp = bprow + ks * 32 + hh * 8;
      b.u[0] = *(const uint4*)(bp);
      b.u[1] = *(const uint4*)(bp + 16);
      acc = __builtin_amdgcn_wmma_f32_16x16x32_f16(false, a.h, false, b.h,
                                                   (short)0, acc, false, false);
    }
    const float bv = bias[n];
#pragma unroll
    for (int r = 0; r < 8; ++r) {           // D: lane holds col n, rows hh*8+r
      float v = acc[r] + bv;
      v = v > 0.f ? v : 0.f;
      out[(hh * 8 + r) * CF + n] = (_Float16)v;
    }
  }
}

// ---------------------------------------------------------------------------
// Edge kernel: one wave = 16 edges. 78 WMMAs per tile.
// ---------------------------------------------------------------------------
__global__ __launch_bounds__(128) void lgeb_edge_kernel(
    const float* __restrict__ x, const int* __restrict__ ei,
    const int* __restrict__ ej, const _Float16* __restrict__ hf,
    const _Float16* __restrict__ We1t, const float* __restrict__ be1,
    const _Float16* __restrict__ We2t, const float* __restrict__ be2,
    const float* __restrict__ Wm, const float* __restrict__ bm,
    const _Float16* __restrict__ Wx1t, const float* __restrict__ bx1,
    const float* __restrict__ Wx2,
    float* __restrict__ wm_acc, float* __restrict__ s_acc,
    float* __restrict__ cnt_acc, int n_edges)
{
  __shared__ __align__(16) _Float16 lds_ein[4][16 * KP_E];
  __shared__ __align__(16) _Float16 lds_m[4][16 * CF];
  __shared__ __align__(16) _Float16 lds_t[4][16 * CF];
  __shared__ int   lds_idx[4][2][16];
  __shared__ float lds_gate[4][16];

  const int wave = threadIdx.x >> 5;
  const int lane = threadIdx.x & 31;
  const int tile = blockIdx.x * 4 + wave;
  if (tile * 16 >= n_edges) return;
  const int e0  = tile * 16;
  const int row = lane & 15;
  const int hh  = lane >> 4;

  _Float16* ein = lds_ein[wave];
  _Float16* mb  = lds_m[wave];
  _Float16* tb  = lds_t[wave];

  // --- indices + Minkowski invariants + zero pad (lanes 0..15) ---
  if (lane < 16) {
    const int i = ei[e0 + lane], j = ej[e0 + lane];
    lds_idx[wave][0][lane] = i;
    lds_idx[wave][1][lane] = j;
    const float4 xi = *(const float4*)(x + 4 * (size_t)i);
    const float4 xj = *(const float4*)(x + 4 * (size_t)j);
    const float d0 = xi.x - xj.x, d1 = xi.y - xj.y,
                d2 = xi.z - xj.z, d3 = xi.w - xj.w;
    const float norms = d0 * d0 - d1 * d1 - d2 * d2 - d3 * d3;
    const float prods = xi.x * xj.x - xi.y * xj.y - xi.z * xj.z - xi.w * xj.w;
    _Float16* r = ein + lane * KP_E;
    r[192] = (_Float16)psi_f(norms);
    r[193] = (_Float16)psi_f(prods);
    unsigned int* zp = (unsigned int*)(r + 194);   // halves 194..223
#pragma unroll
    for (int k = 0; k < 15; ++k) zp[k] = 0u;
  }
  lds_sync();

  // --- gather h_i, h_j rows (each lane copies 24 halves of each row, from half 48*hh) ---
  {
    const int i = lds_idx[wave][0][row];
    const int j = lds_idx[wave][1][row];
    const uint4* si = (const uint4*)(hf + (size_t)i * CF + hh * 48);
    const uint4* sj = (const uint4*)(hf + (size_t)j * CF + hh * 48);
    uint4* di = (uint4*)(ein + row * KP_E + hh * 48);
    uint4* dj = (uint4*)(ein + row * KP_E + CF + hh * 48);
#pragma unroll
    for (int c = 0; c < 3; ++c) { di[c] = si[c]; dj[c] = sj[c]; }
  }
  lds_sync();

  // --- m = relu(relu(e_in @ We1 + be1) @ We2 + be2) ---
  wave_gemm_relu_lds<7>(ein, KP_E, We1t, KP_E, be1, tb, lane);
  lds_sync();
  wave_gemm_relu_lds<3>(tb, CF, We2t, CF, be2, mb, lane);
  lds_sync();

  // --- gate = sigmoid(m @ Wm + bm) ---
  if (lane < 16) {
    float g = bm[0];
    for (int c = 0; c < CF; ++c) g += (float)mb[lane * CF + c] * Wm[c];
    lds_gate[wave][lane] = 1.f / (1.f + __expf(-g));
  }
  // --- t2 = relu(m @ Wx1 + bx1) ---
  wave_gemm_relu_lds<3>(mb, CF, Wx1t, CF, bx1, tb, lane);
  lds_sync();

  // --- scatter: phi_x * x_j, count ---
  if (lane < 16) {
    float p = 0.f;
    for (int c = 0; c < CF; ++c) p += (float)tb[lane * CF + c] * Wx2[c];
    const int i = lds_idx[wave][0][lane];
    const int j = lds_idx[wave][1][lane];
    const float4 xj = *(const float4*)(x + 4 * (size_t)j);
    atomicAdd(s_acc + 4 * (size_t)i + 0, p * xj.x);
    atomicAdd(s_acc + 4 * (size_t)i + 1, p * xj.y);
    atomicAdd(s_acc + 4 * (size_t)i + 2, p * xj.z);
    atomicAdd(s_acc + 4 * (size_t)i + 3, p * xj.w);
    atomicAdd(cnt_acc + i, 1.f);
  }
  // --- scatter: gate * m (each lane: one row, 48 channels) ---
  {
    const int   i = lds_idx[wave][0][row];
    const float g = lds_gate[wave][row];
    const int  c0 = hh * 48;
#pragma unroll 8
    for (int c = 0; c < 48; ++c)
      atomicAdd(wm_acc + (size_t)i * CF + c0 + c,
                g * (float)mb[row * CF + c0 + c]);
  }
}

// ---------------------------------------------------------------------------
// Node kernel: one wave = 16 nodes. 54 WMMAs per tile + x update.
// ---------------------------------------------------------------------------
__global__ __launch_bounds__(128) void lgeb_node_kernel(
    const float* __restrict__ x, const float* __restrict__ h,
    const _Float16* __restrict__ hf, const float* __restrict__ wm_acc,
    const float* __restrict__ s_acc, const float* __restrict__ cnt_acc,
    const _Float16* __restrict__ Wh1t, const float* __restrict__ bh1,
    const _Float16* __restrict__ Wh2t, const float* __restrict__ bh2,
    float* __restrict__ h_out, float* __restrict__ x_out, int n_nodes)
{
  __shared__ __align__(16) _Float16 lds_a[4][16 * D_H];
  __shared__ __align__(16) _Float16 lds_t[4][16 * CF];

  const int wave = threadIdx.x >> 5;
  const int lane = threadIdx.x & 31;
  const int tile = blockIdx.x * 4 + wave;
  if (tile * 16 >= n_nodes) return;
  const int base = tile * 16;
  const int row  = lane & 15;
  const int hh   = lane >> 4;

  _Float16* ab = lds_a[wave];
  _Float16* tb = lds_t[wave];

  // --- build A = concat[h_f16, f16(wm)] (each lane: 24 halves of h, 48 of wm, from 48*hh) ---
  {
    const int node = base + row;
    const uint4* sh = (const uint4*)(hf + (size_t)node * CF + hh * 48);
    uint4* dh = (uint4*)(ab + row * D_H + hh * 48);
#pragma unroll
    for (int c = 0; c < 3; ++c) dh[c] = sh[c];
    const float* sw = wm_acc + (size_t)node * CF + hh * 48;
    _Float16* dw = ab + row * D_H + CF + hh * 48;
#pragma unroll 8
    for (int c = 0; c < 48; ++c) dw[c] = (_Float16)sw[c];
  }
  lds_sync();

  // --- layer 1: relu(A @ Wh1 + bh1) -> LDS ---
  wave_gemm_relu_lds<6>(ab, D_H, Wh1t, D_H, bh1, tb, lane);
  lds_sync();

  // --- layer 2 + residual, f32 out to global ---
#pragma unroll
  for (int nt = 0; nt < 6; ++nt) {
    v8f acc = {};
    const int n = nt * 16 + row;
    const _Float16* bprow = Wh2t + (size_t)n * CF;
#pragma unroll
    for (int ks = 0; ks < 3; ++ks) {
      Frag a, b;
      const _Float16* ap = tb + row * CF + ks * 32 + hh * 8;
      a.u[0] = *(const uint4*)(ap);
      a.u[1] = *(const uint4*)(ap + 16);
      const _Float16* bp = bprow + ks * 32 + hh * 8;
      b.u[0] = *(const uint4*)(bp);
      b.u[1] = *(const uint4*)(bp + 16);
      acc = __builtin_amdgcn_wmma_f32_16x16x32_f16(false, a.h, false, b.h,
                                                   (short)0, acc, false, false);
    }
    const float bv = bh2[n];
#pragma unroll
    for (int r = 0; r < 8; ++r) {
      const size_t node = base + hh * 8 + r;
      h_out[node * CF + n] = h[node * CF + n] + acc[r] + bv;
    }
  }

  // --- x_out = x + C * mean ---
  if (lane < 16) {
    const size_t node = base + lane;
    const float cv = cnt_acc[node];
    const float inv = (cv > 0.f) ? (1.f / fmaxf(cv, 1.f)) : 0.f;
#pragma unroll
    for (int c = 0; c < 4; ++c)
      x_out[node * 4 + c] = x[node * 4 + c] + 0.005f * (s_acc[node * 4 + c] * inv);
  }
}

// ---------------------------------------------------------------------------
// Prep kernels
// ---------------------------------------------------------------------------
__global__ void zero_f32(float* __restrict__ p, int n) {
  for (int i = blockIdx.x * blockDim.x + threadIdx.x; i < n;
       i += gridDim.x * blockDim.x) p[i] = 0.f;
}
__global__ void f32_to_f16(const float* __restrict__ s, _Float16* __restrict__ d,
                           int n) {
  for (int i = blockIdx.x * blockDim.x + threadIdx.x; i < n;
       i += gridDim.x * blockDim.x) d[i] = (_Float16)s[i];
}
// src (K,N) row-major -> dst (N,Kp) f16, zero-padded in K.
__global__ void transpose_pad(const float* __restrict__ src,
                              _Float16* __restrict__ dst, int K, int N, int Kp) {
  const int n_el = N * Kp;
  for (int i = blockIdx.x * blockDim.x + threadIdx.x; i < n_el;
       i += gridDim.x * blockDim.x) {
    const int n = i / Kp, k = i - n * Kp;
    dst[i] = (k < K) ? (_Float16)src[k * N + n] : (_Float16)0.f;
  }
}

// ---------------------------------------------------------------------------
extern "C" void kernel_launch(void* const* d_in, const int* in_sizes, int n_in,
                              void* d_out, int out_size, void* d_ws, size_t ws_size,
                              hipStream_t stream) {
  const float* x   = (const float*)d_in[0];
  const float* h   = (const float*)d_in[1];
  const int*   ei  = (const int*)d_in[2];
  const int*   ej  = (const int*)d_in[3];
  const float* We1 = (const float*)d_in[4];
  const float* be1 = (const float*)d_in[5];
  const float* We2 = (const float*)d_in[6];
  const float* be2 = (const float*)d_in[7];
  const float* Wm  = (const float*)d_in[8];
  const float* bm  = (const float*)d_in[9];
  const float* Wh1 = (const float*)d_in[10];
  const float* bh1 = (const float*)d_in[11];
  const float* Wh2 = (const float*)d_in[12];
  const float* bh2 = (const float*)d_in[13];
  const float* Wx1 = (const float*)d_in[14];
  const float* bx1 = (const float*)d_in[15];
  const float* Wx2 = (const float*)d_in[16];

  const int n_nodes = in_sizes[0] / 4;
  const int n_edges = in_sizes[2];

  // Workspace layout (16B-aligned segments; ~29 MB total).
  char* ws = (char*)d_ws;
  size_t off = 0;
  float* wm_acc = (float*)(ws + off);  off += (size_t)n_nodes * CF * 4;
  float* s_acc  = (float*)(ws + off);  off += (size_t)n_nodes * 4 * 4;
  float* cnt    = (float*)(ws + off);  off += (size_t)n_nodes * 4;
  off = (off + 15) & ~(size_t)15;
  _Float16* hf   = (_Float16*)(ws + off); off += (size_t)n_nodes * CF * 2;
  off = (off + 15) & ~(size_t)15;
  _Float16* We1t = (_Float16*)(ws + off); off += (size_t)CF * KP_E * 2;
  _Float16* We2t = (_Float16*)(ws + off); off += (size_t)CF * CF * 2;
  _Float16* Wx1t = (_Float16*)(ws + off); off += (size_t)CF * CF * 2;
  _Float16* Wh1t = (_Float16*)(ws + off); off += (size_t)CF * D_H * 2;
  _Float16* Wh2t = (_Float16*)(ws + off); off += (size_t)CF * CF * 2;
  (void)ws_size; (void)n_in; (void)out_size;

  float* h_out = (float*)d_out;
  float* x_out = (float*)d_out + (size_t)n_nodes * CF;

  // Prep: zero accumulators (wm|s|cnt contiguous), f16 copies of h + weights.
  zero_f32<<<2048, 256, 0, stream>>>(wm_acc, n_nodes * (CF + 4 + 1));
  f32_to_f16<<<(n_nodes * CF + 255) / 256, 256, 0, stream>>>(h, hf, n_nodes * CF);
  transpose_pad<<<(CF * KP_E + 255) / 256, 256, 0, stream>>>(We1, We1t, 2 * CF + 2, CF, KP_E);
  transpose_pad<<<(CF * CF + 255) / 256, 256, 0, stream>>>(We2, We2t, CF, CF, CF);
  transpose_pad<<<(CF * CF + 255) / 256, 256, 0, stream>>>(Wx1, Wx1t, CF, CF, CF);
  transpose_pad<<<(CF * D_H + 255) / 256, 256, 0, stream>>>(Wh1, Wh1t, D_H, CF, D_H);
  transpose_pad<<<(CF * CF + 255) / 256, 256, 0, stream>>>(Wh2, Wh2t, CF, CF, CF);

  // Edge phase: 16 edges per wave, 4 waves per block.
  const int e_tiles = (n_edges + 15) / 16;
  lgeb_edge_kernel<<<(e_tiles + 3) / 4, 128, 0, stream>>>(
      x, ei, ej, hf, We1t, be1, We2t, be2, Wm, bm, Wx1t, bx1, Wx2,
      wm_acc, s_acc, cnt, n_edges);

  // Node phase: 16 nodes per wave, 4 waves per block.
  const int n_tiles = (n_nodes + 15) / 16;
  lgeb_node_kernel<<<(n_tiles + 3) / 4, 128, 0, stream>>>(
      x, h, hf, wm_acc, s_acc, cnt, Wh1t, bh1, Wh2t, bh2, h_out, x_out, n_nodes);
}
